// MaskedCrossAttention_21663815041181
// MI455X (gfx1250) — compile-verified, matrix operand order checked by reading, not run on hardware
//
#include <hip/hip_runtime.h>
#include <hip/hip_bf16.h>

typedef __attribute__((ext_vector_type(16))) __bf16 v16bf;
typedef __attribute__((ext_vector_type(8)))  float  v8f;

#define WMMA_BF16(a, b, c) \
  __builtin_amdgcn_wmma_f32_16x16x32_bf16(false, (a), false, (b), (short)0, (c), false, false)

// ---------------------------------------------------------------------------
// LayerNorm over rows of 1024, output bf16. One block (256 thr) per row.
// ---------------------------------------------------------------------------
__global__ __launch_bounds__(256) void ln_kernel(const float* __restrict__ x,
                                                 const float* __restrict__ gamma,
                                                 const float* __restrict__ beta,
                                                 __bf16* __restrict__ out) {
  __shared__ float red[256];
  const int row = blockIdx.x;
  const float* xr = x + (size_t)row * 1024;
  float v4[4];
  float s = 0.f;
#pragma unroll
  for (int i = 0; i < 4; ++i) { v4[i] = xr[threadIdx.x + i * 256]; s += v4[i]; }
  red[threadIdx.x] = s;
  __syncthreads();
  for (int o = 128; o > 0; o >>= 1) {
    if (threadIdx.x < o) red[threadIdx.x] += red[threadIdx.x + o];
    __syncthreads();
  }
  const float mu = red[0] * (1.f / 1024.f);
  __syncthreads();
  float vs = 0.f;
#pragma unroll
  for (int i = 0; i < 4; ++i) { float d = v4[i] - mu; vs += d * d; }
  red[threadIdx.x] = vs;
  __syncthreads();
  for (int o = 128; o > 0; o >>= 1) {
    if (threadIdx.x < o) red[threadIdx.x] += red[threadIdx.x + o];
    __syncthreads();
  }
  const float rstd = rsqrtf(red[0] * (1.f / 1024.f) + 1e-5f);
#pragma unroll
  for (int i = 0; i < 4; ++i) {
    const int c = threadIdx.x + i * 256;
    out[(size_t)row * 1024 + c] = (__bf16)((v4[i] - mu) * rstd * gamma[c] + beta[c]);
  }
}

// ---------------------------------------------------------------------------
// Transpose+convert weights: in f32 [K][N] row-major -> out bf16 [N][K].
// ---------------------------------------------------------------------------
__global__ __launch_bounds__(256) void transpose_cvt(const float* __restrict__ in,
                                                     __bf16* __restrict__ out,
                                                     int K, int N) {
  const int idx = blockIdx.x * 256 + threadIdx.x;
  if (idx >= K * N) return;
  const int n = idx / K;
  const int kk = idx - n * K;
  out[idx] = (__bf16)in[(size_t)kk * N + n];
}

// ---------------------------------------------------------------------------
// Straight f32 -> bf16 convert.
// ---------------------------------------------------------------------------
__global__ __launch_bounds__(256) void cvt_bf16(const float* __restrict__ in,
                                                __bf16* __restrict__ out, int n) {
  const int i = blockIdx.x * 256 + threadIdx.x;
  if (i < n) out[i] = (__bf16)in[i];
}

// ---------------------------------------------------------------------------
// Inclusive cumsum of locations per batch (n=2048, serial; tiny cost).
// ---------------------------------------------------------------------------
__global__ void cumsum_kernel(const int* __restrict__ loc, int* __restrict__ tt, int n) {
  if (threadIdx.x != 0) return;
  const int* l = loc + (size_t)blockIdx.x * n;
  int* o = tt + (size_t)blockIdx.x * n;
  int s = 0;
  for (int i = 0; i < n; ++i) { s += l[i]; o[i] = s; }
}

// ---------------------------------------------------------------------------
// Generic TN GEMM: C[M][N] (+epilogue) = A[M][K] (row-major bf16) x
// Bt[N][K] (row-major bf16).  Block = 128 thr = 4 waves; block tile 128x64.
// Each wave computes a 32x64 slab: 2 A frags x 4 B frags -> 8 WMMA per
// 32-wide k-step (1.5 vmem instr / WMMA, B frags reused across 2 M tiles).
// mode 0: bf16 row-major; mode 1: bf16 batch-transposed (vT); mode 2: f32.
// ---------------------------------------------------------------------------
__global__ __launch_bounds__(128) void gemm_tn_bf16(const __bf16* __restrict__ A,
                                                    const __bf16* __restrict__ Bt,
                                                    void* __restrict__ Cout,
                                                    int N, int K, int lda, int ldb,
                                                    int mode, int tRows) {
  const int wave   = threadIdx.x >> 5;
  const int lane   = threadIdx.x & 31;
  const int lane15 = lane & 15;
  const int khalf  = (lane & 16) ? 16 : 0;
  const int m0 = blockIdx.x * 128 + wave * 32;
  const int n0 = blockIdx.y * 64;

  v8f acc[2][4] = {};
  const __bf16* aPtr0 = A  + (size_t)(m0 + lane15) * lda + khalf;
  const __bf16* aPtr1 = A  + (size_t)(m0 + 16 + lane15) * lda + khalf;
  const __bf16* bPtr  = Bt + (size_t)(n0 + lane15) * ldb + khalf;

  for (int k0 = 0; k0 < K; k0 += 32) {
    v16bf a0 = *(const v16bf*)(aPtr0 + k0);
    v16bf a1 = *(const v16bf*)(aPtr1 + k0);
#pragma unroll
    for (int nt = 0; nt < 4; ++nt) {
      v16bf b = *(const v16bf*)(bPtr + (size_t)(nt * 16) * ldb + k0);
      acc[0][nt] = WMMA_BF16(a0, b, acc[0][nt]);
      acc[1][nt] = WMMA_BF16(a1, b, acc[1][nt]);
    }
  }

#pragma unroll
  for (int mi = 0; mi < 2; ++mi) {
    const int mrow = m0 + mi * 16 + ((lane & 16) ? 8 : 0);
#pragma unroll
    for (int nt = 0; nt < 4; ++nt) {
      const int col = n0 + nt * 16 + lane15;
#pragma unroll
      for (int r = 0; r < 8; ++r) {
        const float v = acc[mi][nt][r];
        const int row = mrow + r;
        if (mode == 0) {
          ((__bf16*)Cout)[(size_t)row * N + col] = (__bf16)v;
        } else if (mode == 1) {
          const int bb = row / tRows;
          const int rr = row - bb * tRows;
          ((__bf16*)Cout)[((size_t)(bb * N + col)) * tRows + rr] = (__bf16)v;
        } else {
          ((float*)Cout)[(size_t)row * N + col] = v;
        }
      }
    }
  }
}

// ---------------------------------------------------------------------------
// Fused masked attention for one (b, h, 16-row text tile).
// Block = 128 thr = 4 waves. Wave w handles key chunk [256w, 256w+256)
// whose viz_time == w+1 (mask is equality with txt_time).
// Scores (f32, 16x1028 padded) + normalized P (bf16, 16x1032 padded) in LDS.
// ---------------------------------------------------------------------------
#define SC_LD 1028
#define P_LD  1032
#define ATTN_SMEM (16 * SC_LD * 4 + 16 * P_LD * 2)

__global__ __launch_bounds__(128) void attn_kernel(const __bf16* __restrict__ q,
                                                   const __bf16* __restrict__ k,
                                                   const __bf16* __restrict__ vt,
                                                   const int* __restrict__ txt_time,
                                                   __bf16* __restrict__ out) {
  extern __shared__ char smem[];
  float*  sc = (float*)smem;                       // [16][SC_LD]
  __bf16* pl = (__bf16*)(smem + 16 * SC_LD * 4);   // [16][P_LD]

  const int i0 = blockIdx.x * 16;
  const int h  = blockIdx.y;
  const int b  = blockIdx.z;
  const int wave   = threadIdx.x >> 5;
  const int lane   = threadIdx.x & 31;
  const int lane15 = lane & 15;
  const int khalf  = (lane & 16) ? 16 : 0;
  const int mhalf  = (lane & 16) ? 8 : 0;

  // Q fragments for this 16-row tile (K = DH = 64 -> two 16x16x32 A frags)
  const __bf16* qb = q + ((size_t)(b * 2048 + i0 + lane15)) * 1024 + h * 64;
  const v16bf aq0 = *(const v16bf*)(qb + khalf);
  const v16bf aq1 = *(const v16bf*)(qb + 32 + khalf);

  int ttr[8];
  const int* ttp = txt_time + b * 2048 + i0;
#pragma unroll
  for (int r = 0; r < 8; ++r) ttr[r] = ttp[r + mhalf];

  // ---- Phase 1: scores = scale * Q.K^T with equality mask -> LDS ----
  const __bf16* kb = k + ((size_t)(b * 1024)) * 1024 + h * 64;
  for (int nt = 0; nt < 16; ++nt) {
    const int j0 = wave * 256 + nt * 16;
    const __bf16* kbp = kb + (size_t)(j0 + lane15) * 1024;
    v16bf b0 = *(const v16bf*)(kbp + khalf);
    v16bf b1 = *(const v16bf*)(kbp + 32 + khalf);
    v8f c = {};
    c = WMMA_BF16(aq0, b0, c);
    c = WMMA_BF16(aq1, b1, c);
    const int col = j0 + lane15;
#pragma unroll
    for (int r = 0; r < 8; ++r) {
      const bool ok = (ttr[r] == wave + 1);   // viz_time of this chunk
      sc[(r + mhalf) * SC_LD + col] = ok ? c[r] * 0.125f : -3.402823466e38f;
    }
  }
  __syncthreads();

  // ---- Phase 2: softmax per row (8 threads per row), zero rows tt==0 ----
  {
    const int row = threadIdx.x >> 3;
    const int sub = threadIdx.x & 7;
    float* sr = sc + row * SC_LD;
    float mx = -3.402823466e38f;
    for (int c = sub; c < 1024; c += 8) mx = fmaxf(mx, sr[c]);
    for (int o = 1; o < 8; o <<= 1) mx = fmaxf(mx, __shfl_xor(mx, o, 32));
    float sum = 0.f;
    for (int c = sub; c < 1024; c += 8) {
      const float e = __expf(sr[c] - mx);
      sr[c] = e;
      sum += e;
    }
    for (int o = 1; o < 8; o <<= 1) sum += __shfl_xor(sum, o, 32);
    const int tt = txt_time[b * 2048 + i0 + row];
    const float inv = (tt == 0) ? 0.f : (1.f / sum);
    for (int c = sub; c < 1024; c += 8) pl[row * P_LD + c] = (__bf16)(sr[c] * inv);
  }
  __syncthreads();

  // ---- Phase 3: out = P @ V ; wave w owns dh tile [16w,16w+16) ----
  const __bf16* vb = vt + ((size_t)(b * 1024 + h * 64 + wave * 16 + lane15)) * 1024 + khalf;
  v8f acc = {};
  for (int kk = 0; kk < 1024; kk += 32) {
    v16bf a  = *(const v16bf*)(pl + (size_t)lane15 * P_LD + kk + khalf);
    v16bf bv = *(const v16bf*)(vb + kk);
    acc = WMMA_BF16(a, bv, acc);
  }
  __bf16* ob = out + ((size_t)(b * 2048 + i0 + mhalf)) * 1024 + h * 64 + wave * 16 + lane15;
#pragma unroll
  for (int r = 0; r < 8; ++r) ob[(size_t)r * 1024] = (__bf16)acc[r];
}

// ---------------------------------------------------------------------------
// Host launcher
// ---------------------------------------------------------------------------
extern "C" void kernel_launch(void* const* d_in, const int* in_sizes, int n_in,
                              void* d_out, int out_size, void* d_ws, size_t ws_size,
                              hipStream_t stream) {
  const float* text      = (const float*)d_in[0];   // (2,2048,1024)
  const float* image     = (const float*)d_in[1];   // (2,4,256,1024)
  const int*   locations = (const int*)  d_in[2];   // (2,2048)
  const float* Wq        = (const float*)d_in[3];   // (1024,1024)
  const float* Wkv       = (const float*)d_in[4];   // (1024,2048)
  const float* Wo        = (const float*)d_in[5];   // (1024,1024)
  const float* gamma     = (const float*)d_in[6];
  const float* beta      = (const float*)d_in[7];
  float* out = (float*)d_out;                        // (2,2048,1024) f32

  char* ws = (char*)d_ws;
  size_t off = 0;
  __bf16* WqT  = (__bf16*)(ws + off); off += (size_t)1024 * 1024 * 2;  // [out][in]
  __bf16* WkvT = (__bf16*)(ws + off); off += (size_t)2048 * 1024 * 2;  // [out][in]
  __bf16* WoT  = (__bf16*)(ws + off); off += (size_t)1024 * 1024 * 2;  // [out][in]
  __bf16* imgB = (__bf16*)(ws + off); off += (size_t)2048 * 1024 * 2;  // [b*tm][d]
  __bf16* tnB  = (__bf16*)(ws + off); off += (size_t)4096 * 1024 * 2;  // [b*n][d]
  __bf16* qB   = (__bf16*)(ws + off); off += (size_t)4096 * 1024 * 2;  // [b*n][h*dh]
  __bf16* kB   = (__bf16*)(ws + off); off += (size_t)2048 * 1024 * 2;  // [b*tm][h*dh]
  __bf16* vtB  = (__bf16*)(ws + off); off += (size_t)2048 * 1024 * 2;  // [b][h*dh][tm]
  __bf16* aoB  = (__bf16*)(ws + off); off += (size_t)4096 * 1024 * 2;  // [b*n][h*dh]
  int*    ttW  = (int*)   (ws + off); off += (size_t)4096 * 4;
  (void)in_sizes; (void)n_in; (void)out_size; (void)ws_size;

  // Prep: weight transposes, image convert, LN, cumsum
  transpose_cvt<<<dim3(1024 * 1024 / 256), 256, 0, stream>>>(Wq,  WqT,  1024, 1024);
  transpose_cvt<<<dim3(2048 * 1024 / 256), 256, 0, stream>>>(Wkv, WkvT, 1024, 2048);
  transpose_cvt<<<dim3(1024 * 1024 / 256), 256, 0, stream>>>(Wo,  WoT,  1024, 1024);
  cvt_bf16<<<dim3(2048 * 1024 / 256), 256, 0, stream>>>(image, imgB, 2048 * 1024);
  ln_kernel<<<dim3(4096), 256, 0, stream>>>(text, gamma, beta, tnB);
  cumsum_kernel<<<dim3(2), 32, 0, stream>>>(locations, ttW, 2048);

  // Projections (TN WMMA GEMMs), block tile 128x64
  // q = tn @ Wq        : M=4096, N=1024, K=1024 -> bf16 row-major
  gemm_tn_bf16<<<dim3(32, 16), 128, 0, stream>>>(tnB, WqT, (void*)qB,
                                                 1024, 1024, 1024, 1024, 0, 0);
  // k = img @ Wkv[:, :1024]  -> bf16 row-major [b*tm][h*dh]
  gemm_tn_bf16<<<dim3(16, 16), 128, 0, stream>>>(imgB, WkvT, (void*)kB,
                                                 1024, 1024, 1024, 1024, 0, 0);
  // v = img @ Wkv[:, 1024:]  -> bf16 transposed per batch: vT[b][h*dh][tm]
  gemm_tn_bf16<<<dim3(16, 16), 128, 0, stream>>>(imgB, WkvT + (size_t)1024 * 1024,
                                                 (void*)vtB, 1024, 1024, 1024, 1024,
                                                 1, 1024);

  // Fused masked attention: grid (n/16, H, b)
  attn_kernel<<<dim3(128, 16, 2), 128, ATTN_SMEM, stream>>>(qB, kB, vtB, ttW, aoB);

  // Final projection: out = attn_out @ Wo -> f32
  gemm_tn_bf16<<<dim3(32, 16), 128, 0, stream>>>(aoB, WoT, (void*)out,
                                                 1024, 1024, 1024, 1024, 2, 0);
}
